// Generator_31937376813206
// MI455X (gfx1250) — matrix operand order repaired and checked by reading, not compiled, not run on hardware
//
#include <hip/hip_runtime.h>
#include <math.h>

// ---------------- constants matching the reference ----------------
#define LATENT     128
#define NEV        8
#define ROWS       32          // B * N_EVENTS
#define FRAMES     64
#define NS         16384
#define NHARM      65          // k = 1..65 (N_HARM+1)
#define OUTC       70
#define PCOLS      (OUTC * FRAMES)   // 4480
#define NOISE_WS   512
#define NOISE_STEP 256
#define SR_F       22050.0f
#define NYQ_F      11025.0f
#define TWO_PI_F   6.28318530717958647692f

typedef __attribute__((ext_vector_type(16))) _Float16 v16h;
typedef __attribute__((ext_vector_type(8)))  float    v8f;

__device__ __forceinline__ int imin(int a, int b) { return a < b ? a : b; }

// ---------------- WMMA fragment builders (CDNA5 wave32 layouts) ----------------
// A: 16x32 f16 (MxK).  lane 0-15 -> M=lane, lanes 16-31 -> M=lane-16.
// VGPR v (as f16 pair a[2v],a[2v+1]):  k = (v>>2)*16 + (lane>=16)*8 + (v&3)*2 + {0,1}
__device__ __forceinline__ v16h load_a_frag(const float* act /*ROWSxLATENT*/,
                                            int mt, int kk, int lane) {
  const int half = lane >> 4;
  const int m    = (lane & 15) + mt * 16;
  v16h a;
#pragma unroll
  for (int v = 0; v < 8; ++v) {
    const int k = kk * 32 + ((v >> 2) << 4) + (half << 3) + ((v & 3) << 1);
    a[2 * v]     = (_Float16)act[m * LATENT + k];
    a[2 * v + 1] = (_Float16)act[m * LATENT + k + 1];
  }
  return a;
}

// B: 32x16 f16 (KxN), row-major f32 source with leading dim `ld`.
// lane 0-15 -> N=lane, lanes 16-31 -> N=lane-16. VGPR v: k = (v>>2)*16 + (lane>=16)*8 + (v&3)*2 + {0,1}
__device__ __forceinline__ v16h load_b_frag(const float* W, int ld,
                                            int nt, int kk, int lane) {
  const int half = lane >> 4;
  const int n    = (lane & 15) + nt * 16;
  v16h b;
#pragma unroll
  for (int v = 0; v < 8; ++v) {
    const int k = kk * 32 + ((v >> 2) << 4) + (half << 3) + ((v & 3) << 1);
    b[2 * v]     = (_Float16)W[k * ld + n];
    b[2 * v + 1] = (_Float16)W[(k + 1) * ld + n];
  }
  return b;
}

__device__ __forceinline__ v8f wmma_step(v16h a, v16h b, v8f c) {
  // D = A*B + C ; emits v_wmma_f32_16x16x32_f16
  return __builtin_amdgcn_wmma_f32_16x16x32_f16(false, a, false, b,
                                                (short)0, c, false, false);
}

__device__ __forceinline__ float lrelu(float v) { return v > 0.f ? v : 0.2f * v; }

// ================= kernel A: ln layer + base stack (3) + net stack (5) =================
// One workgroup, 8 waves; activations (32x128) live in LDS; each wave owns one
// 16-col N tile, loops mt=0..1, K=128 in 4 WMMA steps.
__global__ __launch_bounds__(256)
void mlp_kernel(const float* __restrict__ x,     const float* __restrict__ ln_w,
                const float* __restrict__ ln_b,  const float* __restrict__ net_ws,
                const float* __restrict__ net_bs,const float* __restrict__ base_ws,
                const float* __restrict__ base_bs,const float* __restrict__ base_wo,
                const float* __restrict__ base_bo,
                float* __restrict__ ws_h5, float* __restrict__ ws_base) {
  __shared__ float shH[ROWS * LATENT];   // h after ln layer (kept)
  __shared__ float shA[ROWS * LATENT];
  __shared__ float shB[ROWS * LATENT];
  const int tid = threadIdx.x, lane = tid & 31, wave = tid >> 5;

  // ---- h = x @ ln_w + ln_b  (rows pick per-event weight column slices) ----
  for (int o = tid * 16; o < tid * 16 + 16; ++o) {
    const int r = o >> 7, c = o & 127;
    const int b = r >> 3, e = r & 7;
    const float* xr = x + b * LATENT;
    const float* wc = ln_w + e * LATENT + c;
    float acc = ln_b[e * LATENT + c];
#pragma unroll 4
    for (int k = 0; k < LATENT; ++k) acc += xr[k] * wc[k * (LATENT * NEV)];
    shH[r * LATENT + c] = acc;
  }
  __syncthreads();

  // ---- base stack: 3 hidden layers (WMMA) ----
  const float* src = shH;
  float* dst = shA;
  for (int l = 0; l < 3; ++l) {
    const float* W  = base_ws + l * LATENT * LATENT;
    const float bv  = base_bs[l * LATENT + (lane & 15) + wave * 16];
    for (int mt = 0; mt < 2; ++mt) {
      v8f acc = {};
#pragma unroll
      for (int kk = 0; kk < 4; ++kk)
        acc = wmma_step(load_a_frag(src, mt, kk, lane),
                        load_b_frag(W, LATENT, wave, kk, lane), acc);
      const int n = (lane & 15) + wave * 16, half = lane >> 4;
#pragma unroll
      for (int rr = 0; rr < 8; ++rr)
        dst[(mt * 16 + rr + half * 8) * LATENT + n] = lrelu(acc[rr] + bv);
    }
    __syncthreads();
    src = dst;
    dst = (dst == shA) ? shB : shA;
  }
  // baselines: hb @ base_wo + base_bo (src == shA after 3 layers)
  if (tid < ROWS) {
    float acc = base_bo[0];
#pragma unroll 4
    for (int k = 0; k < LATENT; ++k) acc += src[tid * LATENT + k] * base_wo[k];
    ws_base[tid] = acc;
  }
  __syncthreads();

  // ---- net stack: 5 hidden layers (WMMA), from shH ----
  const float* s2 = shH;
  float* d2 = shB;
  for (int l = 0; l < 5; ++l) {
    const float* W = net_ws + l * LATENT * LATENT;
    const float bv = net_bs[l * LATENT + (lane & 15) + wave * 16];
    for (int mt = 0; mt < 2; ++mt) {
      v8f acc = {};
#pragma unroll
      for (int kk = 0; kk < 4; ++kk)
        acc = wmma_step(load_a_frag(s2, mt, kk, lane),
                        load_b_frag(W, LATENT, wave, kk, lane), acc);
      const int n = (lane & 15) + wave * 16, half = lane >> 4;
#pragma unroll
      for (int rr = 0; rr < 8; ++rr)
        d2[(mt * 16 + rr + half * 8) * LATENT + n] = lrelu(acc[rr] + bv);
    }
    __syncthreads();
    s2 = d2;
    d2 = (d2 == shA) ? shB : shA;
  }
  // after 5 layers result sits in shB (s2)
  for (int i = tid; i < ROWS * LATENT; i += 256) ws_h5[i] = s2[i];
}

// ================= kernel B: p = (h5 @ net_wo + net_bo + 1)/2  (32 x 4480) =================
// 70 blocks x 8 waves; each wave one 16x16 tile (280 N tiles x 2 M tiles).
__global__ __launch_bounds__(256)
void pgen_kernel(const float* __restrict__ ws_h5, const float* __restrict__ net_wo,
                 const float* __restrict__ net_bo, float* __restrict__ ws_p) {
  __shared__ float shH[ROWS * LATENT];
  const int tid = threadIdx.x, lane = tid & 31, wave = tid >> 5;
  for (int i = tid; i < ROWS * LATENT; i += 256) shH[i] = ws_h5[i];
  __syncthreads();
  const int nt = blockIdx.x * 4 + (wave & 3);
  const int mt = wave >> 2;
  v8f acc = {};
#pragma unroll
  for (int kk = 0; kk < 4; ++kk)
    acc = wmma_step(load_a_frag(shH, mt, kk, lane),
                    load_b_frag(net_wo, PCOLS, nt, kk, lane), acc);
  const int n = (lane & 15) + nt * 16, half = lane >> 4;
  const float bv = net_bo[n];
#pragma unroll
  for (int rr = 0; rr < 8; ++rr) {
    const int m = mt * 16 + rr + half * 8;
    ws_p[m * PCOLS + n] = (acc[rr] + bv + 1.f) * 0.5f;
  }
}

// ================= kernel C: oscillator bank =================
// One block per (b,e) row. Phase via block prefix-sum of upsampled f0,
// 65 harmonics via Chebyshev sine recurrence (1 sincos / sample).
__global__ __launch_bounds__(256)
void osc_kernel(const float* __restrict__ ws_p, float* __restrict__ ws_osc) {
  const int r = blockIdx.x, tid = threadIdx.x;
  __shared__ float f0hz[FRAMES];
  __shared__ float amp[NHARM * FRAMES];
  __shared__ float partial[256];
  const float* p = ws_p + r * PCOLS;
  if (tid < FRAMES) f0hz[tid] = 20.0f + p[0 * FRAMES + tid] * 980.0f;
  for (int i = tid; i < NHARM * FRAMES; i += 256) {
    const int k = i / FRAMES, f = i - k * FRAMES;
    const int ch = (k == 0) ? 1 : (4 + k);          // osc_env then harm_env
    amp[i] = p[ch * FRAMES + f];
  }
  __syncthreads();

  const float posScale = 63.0f / 16383.0f;
  const int n0 = tid * 64;
  float lsum = 0.f;
  for (int i = 0; i < 64; ++i) {
    const float pos = (n0 + i) * posScale;
    const int i0 = (int)pos, i1 = imin(i0 + 1, FRAMES - 1);
    const float w = pos - (float)i0;
    lsum += f0hz[i0] * (1.f - w) + f0hz[i1] * w;
  }
  partial[tid] = lsum;
  __syncthreads();
  for (int off = 1; off < 256; off <<= 1) {      // Hillis-Steele inclusive scan
    const float v   = partial[tid];
    const float add = (tid >= off) ? partial[tid - off] : 0.f;
    __syncthreads();
    partial[tid] = v + add;
    __syncthreads();
  }
  float runSum = (tid > 0) ? partial[tid - 1] : 0.f;

  const float phScale = TWO_PI_F / SR_F;
  float* orow = ws_osc + r * NS;
  for (int i = 0; i < 64; ++i) {
    const int n = n0 + i;
    const float pos = n * posScale;
    const int i0 = (int)pos, i1 = imin(i0 + 1, FRAMES - 1);
    const float w = pos - (float)i0;
    const float f0 = f0hz[i0] * (1.f - w) + f0hz[i1] * w;
    runSum += f0;
    const float phase = runSum * phScale;
    float s, c;
    __sincosf(phase, &s, &c);
    const float twoc = 2.f * c;
    float sPrev = 0.f, sCur = s, acc = 0.f, fk = f0;
#pragma unroll 1
    for (int k = 1; k <= NHARM; ++k) {
      if (fk < NYQ_F) {
        const float a0 = amp[(k - 1) * FRAMES + i0];
        const float a1 = amp[(k - 1) * FRAMES + i1];
        acc += (a0 + (a1 - a0) * w) * sCur;
      }
      const float sNext = twoc * sCur - sPrev;
      sPrev = sCur; sCur = sNext;
      fk += f0;
    }
    orow[n] = acc;
  }
}

// ================= kernel D: spectral-filtered noise, overlap-add =================
// rfft->Gauss filt->irfft == circular conv with g = irfft(filt); g built with
// a cosine recurrence, conv done in LDS. One block per (row, frame).
__global__ __launch_bounds__(256)
void noise_kernel(const float* __restrict__ noise, const float* __restrict__ ws_p,
                  float* __restrict__ ws_noise) {
  const int blk = blockIdx.x, r = blk >> 6, f = blk & 63, tid = threadIdx.x;
  __shared__ float xw[NOISE_WS];
  __shared__ float filt[NOISE_WS / 2 + 1];
  __shared__ float g[NOISE_WS];
  const float* p = ws_p + r * PCOLS;
  const float nstd = p[4 * FRAMES + f] + 1e-4f;
  const float nenv = p[2 * FRAMES + f];
  const int start = f * NOISE_STEP;
  const float winScale = TWO_PI_F / (float)(NOISE_WS - 1);   // np.hanning
  for (int i = tid; i < NOISE_WS; i += 256) {
    const int gi = start + i;
    const float v = (gi < NS) ? noise[r * NS + gi] : 0.f;    // implicit zero-pad
    xw[i] = v * (0.5f - 0.5f * __cosf(winScale * i));
  }
  for (int j = tid; j <= NOISE_WS / 2; j += 256) {
    const float t = ((float)j / (float)(NOISE_WS / 2)) / nstd;
    filt[j] = __expf(-0.5f * t * t);
  }
  __syncthreads();
  const float thScale = TWO_PI_F / (float)NOISE_WS;
  for (int n = tid; n < NOISE_WS; n += 256) {
    const float c1 = __cosf(thScale * n);
    float cPrev = 1.f, cCur = c1, sum = filt[0];
#pragma unroll 1
    for (int j = 1; j <= 255; ++j) {
      sum += 2.f * filt[j] * cCur;
      const float cNext = 2.f * c1 * cCur - cPrev;
      cPrev = cCur; cCur = cNext;
    }
    sum += filt[256] * ((n & 1) ? -1.f : 1.f);               // cos(pi*n)
    g[n] = sum * (1.0f / (float)NOISE_WS);
  }
  __syncthreads();
  for (int n = tid; n < NOISE_WS; n += 256) {
    float acc = 0.f;
#pragma unroll 4
    for (int m = 0; m < NOISE_WS; ++m)
      acc += xw[m] * g[(n - m) & (NOISE_WS - 1)];
    acc *= (0.5f - 0.5f * __cosf(winScale * n)) * nenv;
    const int gi = start + n;
    if (gi < NS) atomicAdd(&ws_noise[r * NS + gi], acc);     // overlap-add
  }
}

// ================= kernel E: (osc+noise)*overall, sum over events =================
__global__ __launch_bounds__(256)
void final_kernel(const float* __restrict__ ws_p, const float* __restrict__ ws_base,
                  const float* __restrict__ ws_osc, const float* __restrict__ ws_noise,
                  float* __restrict__ out) {
  const int idx = blockIdx.x * 256 + threadIdx.x;  // 0..65535
  const int b = idx >> 14, n = idx & (NS - 1);
  const float pos = n * (63.0f / 16383.0f);
  const int i0 = (int)pos, i1 = imin(i0 + 1, FRAMES - 1);
  const float w = pos - (float)i0;
  float acc = 0.f;
#pragma unroll
  for (int e = 0; e < NEV; ++e) {
    const int r = b * NEV + e;
    const float* ov = ws_p + r * PCOLS + 3 * FRAMES;
    const float o = ov[i0] * (1.f - w) + ov[i1] * w + ws_base[r];
    acc += (ws_osc[r * NS + n] + ws_noise[r * NS + n]) * o;
  }
  out[idx] = acc;
}

__global__ void zero_kernel(float* __restrict__ p, int n) {
  const int i = blockIdx.x * 256 + threadIdx.x;
  if (i < n) p[i] = 0.f;
}

// ================= launcher =================
extern "C" void kernel_launch(void* const* d_in, const int* in_sizes, int n_in,
                              void* d_out, int out_size, void* d_ws, size_t ws_size,
                              hipStream_t stream) {
  const float* x       = (const float*)d_in[0];
  const float* noise   = (const float*)d_in[1];
  const float* ln_w    = (const float*)d_in[2];
  const float* ln_b    = (const float*)d_in[3];
  const float* net_ws  = (const float*)d_in[4];
  const float* net_bs  = (const float*)d_in[5];
  const float* net_wo  = (const float*)d_in[6];
  const float* net_bo  = (const float*)d_in[7];
  const float* base_ws = (const float*)d_in[8];
  const float* base_bs = (const float*)d_in[9];
  const float* base_wo = (const float*)d_in[10];
  const float* base_bo = (const float*)d_in[11];

  float* ws       = (float*)d_ws;
  float* ws_h5    = ws;                       // 4096
  float* ws_base  = ws + 4096;                // 32 (padded to 4128)
  float* ws_p     = ws + 4128;                // 143360
  float* ws_osc   = ws_p + ROWS * PCOLS;      // 524288
  float* ws_noise = ws_osc + ROWS * NS;       // 524288
  float* out      = (float*)d_out;

  hipLaunchKernelGGL(zero_kernel, dim3((ROWS * NS + 255) / 256), dim3(256), 0, stream,
                     ws_noise, ROWS * NS);
  hipLaunchKernelGGL(mlp_kernel, dim3(1), dim3(256), 0, stream,
                     x, ln_w, ln_b, net_ws, net_bs, base_ws, base_bs, base_wo, base_bo,
                     ws_h5, ws_base);
  hipLaunchKernelGGL(pgen_kernel, dim3(70), dim3(256), 0, stream,
                     ws_h5, net_wo, net_bo, ws_p);
  hipLaunchKernelGGL(osc_kernel, dim3(ROWS), dim3(256), 0, stream, ws_p, ws_osc);
  hipLaunchKernelGGL(noise_kernel, dim3(ROWS * FRAMES), dim3(256), 0, stream,
                     noise, ws_p, ws_noise);
  hipLaunchKernelGGL(final_kernel, dim3(256), dim3(256), 0, stream,
                     ws_p, ws_base, ws_osc, ws_noise, out);
}
